// DocumentEncoder_1881195675861
// MI455X (gfx1250) — compile-verified, matrix operand order checked by reading, not run on hardware
//
#include <hip/hip_runtime.h>

#define EMB    512
#define BATCH  64
#define SEQ    2048
#define CH     16           // chunks per document for the heavy gather passes
#define CHLEN  (SEQ / CH)   // 128 tokens per chunk

typedef float    v2f  __attribute__((ext_vector_type(2)));
typedef float    v8f  __attribute__((ext_vector_type(8)));
typedef _Float16 v16h __attribute__((ext_vector_type(16)));

// ---------------------------------------------------------------------------
// Kernel 1a: partial bag sums.  grid (BATCH, CH), 128 threads * float4 = 512d.
// Each block streams CHLEN embedding rows (coalesced 2KB reads, L2-resident
// table after warm-up) into partial[b, c, :].
// ---------------------------------------------------------------------------
__global__ void k_bag_partial(const int* __restrict__ tokens,
                              const float* __restrict__ table,
                              float* __restrict__ partial) {
  const int b  = blockIdx.x;
  const int c  = blockIdx.y;
  const int d4 = threadIdx.x;                       // 0..127
  const int* trow = tokens + (size_t)b * SEQ + (size_t)c * CHLEN;
  float4 acc = make_float4(0.f, 0.f, 0.f, 0.f);
  for (int l = 0; l < CHLEN; ++l) {
    const int row = trow[l] + 1;
    const float4 v = *(const float4*)(table + (size_t)row * EMB + d4 * 4);
    acc.x += v.x; acc.y += v.y; acc.z += v.z; acc.w += v.w;
  }
  *(float4*)(partial + ((size_t)b * CH + c) * EMB + d4 * 4) = acc;
}

// Kernel 1b: h[b,:] = (1/SEQ) * sum_c partial[b,c,:]   (deterministic order)
__global__ void k_bag_reduce(const float* __restrict__ partial,
                             float* __restrict__ h) {
  const int b  = blockIdx.x;
  const int d4 = threadIdx.x;
  float4 acc = make_float4(0.f, 0.f, 0.f, 0.f);
#pragma unroll
  for (int c = 0; c < CH; ++c) {
    const float4 v = *(const float4*)(partial + ((size_t)b * CH + c) * EMB + d4 * 4);
    acc.x += v.x; acc.y += v.y; acc.z += v.z; acc.w += v.w;
  }
  const float s = 1.0f / (float)SEQ;
  float4 r = make_float4(acc.x * s, acc.y * s, acc.z * s, acc.w * s);
  *(float4*)(h + (size_t)b * EMB + d4 * 4) = r;
}

// ---------------------------------------------------------------------------
// Kernel 2: u = h @ W_b^T  (M=64, N=512, K=512) via V_WMMA_F32_16X16X4_F32.
// One wave32 per 16x16 output tile; K stepped by 4.
// A layout (16x4 f32, ISA 7.12.2): lanes 0-15 hold M=lane with K = k,k+1;
// lanes 16-31 hold M=lane-16 with K = k+2,k+3. B (4x16) mirrors with N=lane;
// since u[m,n] = sum_k h[m,k]*W[n,k], the B fragment reads W row-wise.
// C/D layout: VGPR j -> row M = j (lanes 0-15) / j+8 (lanes 16-31), N = lane.
// ---------------------------------------------------------------------------
__global__ void k_u_gemm_wmma(const float* __restrict__ h,
                              const float* __restrict__ W,
                              float* __restrict__ u) {
  const int nt   = blockIdx.x;          // 0..31 : 16-wide column tile of u
  const int mt   = blockIdx.y;          // 0..3  : 16-tall row tile
  const int lane = threadIdx.x;         // 0..31 (single wave, EXEC all ones)
  const int half = lane >> 4;
  const int lid  = lane & 15;
  v8f c = {};

  const float* arow = h + (size_t)(mt * 16 + lid) * EMB;   // A[m,k] = h[m,k]
  const float* brow = W + (size_t)(nt * 16 + lid) * EMB;   // B[k,n] = W[n,k]

#if __has_builtin(__builtin_amdgcn_wmma_f32_16x16x4_f32)
  for (int k = 0; k < EMB; k += 4) {
    v2f a, bm;
    a[0]  = arow[k + 2 * half + 0];
    a[1]  = arow[k + 2 * half + 1];
    bm[0] = brow[k + 2 * half + 0];
    bm[1] = brow[k + 2 * half + 1];
    c = __builtin_amdgcn_wmma_f32_16x16x4_f32(false, a, false, bm,
                                              (short)0, c, false, false);
  }
#else
  // Fallback: codegen-confirmed f16 WMMA (K stepped by 32).
  for (int k = 0; k < EMB; k += 32) {
    v16h a, bm;
    const int base = k + 8 * half;
#pragma unroll
    for (int i = 0; i < 8; ++i) {
      a[i]      = (_Float16)arow[base + i];
      a[i + 8]  = (_Float16)arow[base + 16 + i];
      bm[i]     = (_Float16)brow[base + i];
      bm[i + 8] = (_Float16)brow[base + 16 + i];
    }
    c = __builtin_amdgcn_wmma_f32_16x16x32_f16(false, a, false, bm,
                                               (short)0, c, false, false);
  }
#endif

#pragma unroll
  for (int j = 0; j < 8; ++j)
    u[(size_t)(mt * 16 + j + 8 * half) * EMB + nt * 16 + lid] = c[j];
}

// ---------------------------------------------------------------------------
// Kernel 3: scores[b,l] = table[tok+1,:] . u[b,:]
// One wave32 per token (8 waves / 256-thread block); shuffle-xor reduction.
// ---------------------------------------------------------------------------
__global__ void k_scores(const int* __restrict__ tokens,
                         const float* __restrict__ table,
                         const float* __restrict__ u,
                         float* __restrict__ scores) {
  const int wid  = threadIdx.x >> 5;
  const int lane = threadIdx.x & 31;
  const int gid  = blockIdx.x * 8 + wid;     // token index in [0, B*SEQ)
  const int b    = gid >> 11;                // / SEQ
  const int row  = tokens[gid] + 1;
  const float* e  = table + (size_t)row * EMB;
  const float* ub = u + (size_t)b * EMB;
  const int base = lane * 16;                // 512 / 32 lanes
  float acc = 0.f;
#pragma unroll
  for (int i = 0; i < 16; i += 4) {
    const float4 ev = *(const float4*)(e + base + i);
    const float4 uv = *(const float4*)(ub + base + i);
    acc += ev.x * uv.x + ev.y * uv.y + ev.z * uv.z + ev.w * uv.w;
  }
#pragma unroll
  for (int off = 16; off > 0; off >>= 1)
    acc += __shfl_xor(acc, off, 32);
  if (lane == 0) scores[gid] = acc;
}

// ---------------------------------------------------------------------------
// Kernel 4a: per-document softmax stats: stats[b] = {max, sum exp(s-max)}.
// One block per document (tiny: 512KB of scores total).
// ---------------------------------------------------------------------------
__global__ void k_softmax_stats(const float* __restrict__ scores,
                                float* __restrict__ stats) {
  const int b   = blockIdx.x;
  const int tid = threadIdx.x;               // 0..255
  __shared__ float red[256];
  const float* sb = scores + (size_t)b * SEQ;

  float m = -3.0e38f;
  for (int l = tid; l < SEQ; l += 256) m = fmaxf(m, sb[l]);
  red[tid] = m; __syncthreads();
  for (int s = 128; s > 0; s >>= 1) {
    if (tid < s) red[tid] = fmaxf(red[tid], red[tid + s]);
    __syncthreads();
  }
  const float M = red[0];
  __syncthreads();

  float z = 0.f;
  for (int l = tid; l < SEQ; l += 256) z += expf(sb[l] - M);
  red[tid] = z; __syncthreads();
  for (int s = 128; s > 0; s >>= 1) {
    if (tid < s) red[tid] += red[tid + s];
    __syncthreads();
  }
  if (tid == 0) { stats[b * 2 + 0] = M; stats[b * 2 + 1] = red[0]; }
}

// ---------------------------------------------------------------------------
// Kernel 4b: partial weighted sums.  grid (BATCH, CH), 128 threads * float4.
// partial[b,c,:] = sum_{l in chunk} exp(s[b,l]-M[b]) * table[tok+1,:]
// ---------------------------------------------------------------------------
__global__ void k_ct_partial(const int* __restrict__ tokens,
                             const float* __restrict__ table,
                             const float* __restrict__ scores,
                             const float* __restrict__ stats,
                             float* __restrict__ partial) {
  const int b  = blockIdx.x;
  const int c  = blockIdx.y;
  const int d4 = threadIdx.x;
  const float M = stats[b * 2 + 0];
  const int* trow = tokens + (size_t)b * SEQ + (size_t)c * CHLEN;
  const float* sb = scores + (size_t)b * SEQ + (size_t)c * CHLEN;
  float4 acc = make_float4(0.f, 0.f, 0.f, 0.f);
  for (int l = 0; l < CHLEN; ++l) {
    const int row  = trow[l] + 1;
    const float wt = expf(sb[l] - M);
    const float4 v = *(const float4*)(table + (size_t)row * EMB + d4 * 4);
    acc.x += wt * v.x; acc.y += wt * v.y; acc.z += wt * v.z; acc.w += wt * v.w;
  }
  *(float4*)(partial + ((size_t)b * CH + c) * EMB + d4 * 4) = acc;
}

// Kernel 4c: out[b,:] = (1/Z[b]) * sum_c partial[b,c,:]
__global__ void k_ct_reduce(const float* __restrict__ partial,
                            const float* __restrict__ stats,
                            float* __restrict__ out) {
  const int b  = blockIdx.x;
  const int d4 = threadIdx.x;
  const float invZ = 1.0f / stats[b * 2 + 1];
  float4 acc = make_float4(0.f, 0.f, 0.f, 0.f);
#pragma unroll
  for (int c = 0; c < CH; ++c) {
    const float4 v = *(const float4*)(partial + ((size_t)b * CH + c) * EMB + d4 * 4);
    acc.x += v.x; acc.y += v.y; acc.z += v.z; acc.w += v.w;
  }
  float4 r = make_float4(acc.x * invZ, acc.y * invZ, acc.z * invZ, acc.w * invZ);
  *(float4*)(out + (size_t)b * EMB + d4 * 4) = r;
}

// ---------------------------------------------------------------------------
extern "C" void kernel_launch(void* const* d_in, const int* in_sizes, int n_in,
                              void* d_out, int out_size, void* d_ws, size_t ws_size,
                              hipStream_t stream) {
  (void)in_sizes; (void)n_in; (void)out_size; (void)ws_size;
  const int*   tokens = (const int*)d_in[0];    // [64, 2048] int32
  const float* table  = (const float*)d_in[1];  // [50258, 512] f32
  const float* W      = (const float*)d_in[2];  // [512, 512] f32
  // d_in[3] = max_len scalar (compile-time constant SEQ here)
  float* out = (float*)d_out;                   // [64, 512] f32

  float* h       = (float*)d_ws;                 // 64*512            (128 KB)
  float* u       = h + BATCH * EMB;              // 64*512            (128 KB)
  float* scores  = u + BATCH * EMB;              // 64*2048           (512 KB)
  float* stats   = scores + BATCH * SEQ;         // 64*2              (512 B)
  float* partial = stats + BATCH * 2;            // 64*16*512         (2 MB)
  // `partial` is reused by both reduction phases; the stream serializes them.

  k_bag_partial  <<<dim3(BATCH, CH), 128, 0, stream>>>(tokens, table, partial);
  k_bag_reduce   <<<BATCH, 128, 0, stream>>>(partial, h);
  k_u_gemm_wmma  <<<dim3(EMB / 16, BATCH / 16), 32, 0, stream>>>(h, W, u);
  k_scores       <<<(BATCH * SEQ) / 8, 256, 0, stream>>>(tokens, table, u, scores);
  k_softmax_stats<<<BATCH, 256, 0, stream>>>(scores, stats);
  k_ct_partial   <<<dim3(BATCH, CH), 128, 0, stream>>>(tokens, table, scores, stats, partial);
  k_ct_reduce    <<<BATCH, 128, 0, stream>>>(partial, stats, out);
}
